// LSTMModel_83339545411774
// MI455X (gfx1250) — compile-verified, matrix operand order checked by reading, not run on hardware
//
#include <hip/hip_runtime.h>
#include <math.h>

// ---------------- problem constants ----------------
#define BB   128      // batch
#define TT   512      // timesteps
#define FF   512      // input features
#define HH   1024     // hidden
#define G4   (4*HH)   // gate rows
#define AOUT 14       // fc outputs
#define NWG  64       // persistent workgroups: HH/16

// ---------------- workspace layout (bytes) ----------------
static constexpr size_t OFF_CTRL = 0;                                   // 256 B: barrier count/gen
static constexpr size_t OFF_H    = 256;                                 // 2 * BB*HH bf16 = 512 KiB
static constexpr size_t OFF_WIH  = OFF_H   + (size_t)2*BB*HH*2;         // packed W_ih bf16: 4 MiB
static constexpr size_t OFF_WHH  = OFF_WIH + (size_t)G4*FF*2;           // packed W_hh bf16: 8 MiB
static constexpr size_t OFF_XBF  = OFF_WHH + (size_t)G4*HH*2;           // x transposed bf16: 64 MiB

// LDS layout (halfs): [0, 65536)  = W_hh frags: 4 gates * 32 ktiles * 512
//                     [65536, 98304) = W_ih frags: 4 gates * 16 ktiles * 512   => 192 KiB total
#define LDS_WIH_BASE (4 * 32 * 512)
#define LDS_HALFS    (LDS_WIH_BASE + 4 * 16 * 512)

// ---------------- types ----------------
typedef __bf16 v16bf __attribute__((ext_vector_type(16)));
typedef float  v8f   __attribute__((ext_vector_type(8)));
typedef unsigned int u32x4 __attribute__((ext_vector_type(4)));
typedef int          i32x8 __attribute__((ext_vector_type(8)));
typedef int          i32x4 __attribute__((ext_vector_type(4)));

union FragU {
    uint4 q[2];
    v16bf v;
};

__device__ __forceinline__ unsigned short f2bf(float x) {
    unsigned u = __float_as_uint(x);
    unsigned r = (u + 0x7fffu + ((u >> 16) & 1u)) >> 16;
    return (unsigned short)r;
}
__device__ __forceinline__ float bf2f(unsigned short h) {
    return __uint_as_float(((unsigned)h) << 16);
}
__device__ __forceinline__ float fsigmoid(float x) {
    return 1.0f / (1.0f + __expf(-x));
}
__device__ __forceinline__ float ftanh(float x) {
    float e = __expf(-2.0f * x);
    return (1.0f - e) / (1.0f + e);
}

// ---------------- prep: x (B,T,F) f32 -> xbf (T,B,F) bf16 ----------------
__global__ void __launch_bounds__(256) prep_x_kernel(const float* __restrict__ x,
                                                     unsigned short* __restrict__ xbf) {
    size_t idx = (size_t)blockIdx.x * 256 + threadIdx.x;
    if (idx >= (size_t)TT * BB * FF) return;
    int f = (int)(idx % FF);
    int b = (int)((idx / FF) % BB);
    int t = (int)(idx / ((size_t)FF * BB));
    xbf[idx] = f2bf(x[((size_t)b * TT + t) * FF + f]);
}

// ---------------- prep: pack weights into per-lane WMMA B-fragment order ----------------
// Fragment (nt, kt): B tile is K=32 x N=16 of W^T, i.e. W[n][k].
// Lane L: n = nt*16 + (L%16), khalf = L/16.
// halfs j=0..7  -> k = kt*32 +      khalf*8 + j
// halfs j=8..15 -> k = kt*32 + 16 + khalf*8 + (j-8)
// Flat addr: ((nt*nKt + kt)*32 + lane)*16 + j
__global__ void __launch_bounds__(256) pack_w_kernel(const float* __restrict__ Wih,
                                                     const float* __restrict__ Whh,
                                                     unsigned short* __restrict__ Pih,
                                                     unsigned short* __restrict__ Phh) {
    size_t idx = (size_t)blockIdx.x * 256 + threadIdx.x;
    const size_t NIH = (size_t)G4 * FF;
    const size_t NHH = (size_t)G4 * HH;
    const float* W; unsigned short* P; int Kdim; size_t rel;
    if (idx < NIH)            { W = Wih; P = Pih; Kdim = FF; rel = idx; }
    else if (idx < NIH + NHH) { W = Whh; P = Phh; Kdim = HH; rel = idx - NIH; }
    else return;

    int    j    = (int)(rel & 15);
    size_t fl   = rel >> 4;           // frag*32 + lane
    int    lane = (int)(fl & 31);
    size_t frag = fl >> 5;
    int    nKt  = Kdim / 32;
    int    kt   = (int)(frag % nKt);
    int    nt   = (int)(frag / nKt);
    int    n    = nt * 16 + (lane & 15);
    int    kh   = lane >> 4;
    int    k    = kt * 32 + ((j < 8) ? (kh * 8 + j) : (16 + kh * 8 + (j - 8)));
    P[rel] = f2bf(W[(size_t)n * Kdim + k]);
}

// ---------------- grid-wide barrier (sense via generation counter) ----------------
__device__ __forceinline__ void grid_barrier(unsigned* cnt, unsigned* gen) {
    __threadfence();
    __syncthreads();
    if (threadIdx.x == 0) {
        unsigned g = __hip_atomic_load(gen, __ATOMIC_ACQUIRE, __HIP_MEMORY_SCOPE_AGENT);
        unsigned a = __hip_atomic_fetch_add(cnt, 1u, __ATOMIC_ACQ_REL, __HIP_MEMORY_SCOPE_AGENT);
        if (a == NWG - 1) {
            __hip_atomic_store(cnt, 0u, __ATOMIC_RELAXED, __HIP_MEMORY_SCOPE_AGENT);
            __hip_atomic_store(gen, g + 1u, __ATOMIC_RELEASE, __HIP_MEMORY_SCOPE_AGENT);
        } else {
            while (__hip_atomic_load(gen, __ATOMIC_ACQUIRE, __HIP_MEMORY_SCOPE_AGENT) == g) {
                __builtin_amdgcn_s_sleep(1);
            }
        }
    }
    __syncthreads();
}

// ---------------- TDM: 1-D tile load global -> LDS ----------------
// Builds the D# per CDNA5 ISA 8.3/8.4: group0 {count=1, lds_addr, global_addr, type=2},
// group1 {data_size=8B, tensor_dim0=tile_dim0=nelem8, dims1=1, stride0=nelem8}.
// Toolchain here is the 6-arg builtin: (u32x4 g0, i32x8 g1, i32x4, i32x4, i32x8, i32 cpol).
#if defined(__has_builtin)
#if __has_builtin(__builtin_amdgcn_tensor_load_to_lds)
#define HAVE_TDM 1
#endif
#endif

#ifdef HAVE_TDM
__device__ __forceinline__ void tdm_load_1d(const void* gsrc, void* ldst, unsigned nelem8) {
    unsigned long long ga = (unsigned long long)(size_t)gsrc;
    unsigned la = (unsigned)(size_t)ldst;            // low 32 bits of LDS aperture addr = LDS offset
    u32x4 g0;
    g0.x = 1u;                                       // count=1 (user descriptor)
    g0.y = la;                                       // lds_addr  [63:32]
    g0.z = (unsigned)(ga & 0xffffffffu);             // global_addr [95:64]
    g0.w = (unsigned)((ga >> 32) & 0x01ffffffu)      // global_addr [120:96]
         | (2u << 30);                               // type=2 ("image")
    i32x8 g1;
    g1[0] = (int)(3u << 16);                         // data_size=3 (8 bytes); wg_mask=0
    g1[1] = (int)((nelem8 & 0xffffu) << 16);         // tensor_dim0[15:0] in bits 63:48
    g1[2] = (int)(((nelem8 >> 16) & 0xffffu)         // tensor_dim0[31:16]
         | (1u << 16));                              // tensor_dim1 = 1 (bits 95:80)
    g1[3] = (int)((nelem8 & 0xffffu) << 16);         // tile_dim0 in bits 127:112 (dim1 hi16 = 0)
    g1[4] = 1;                                       // tile_dim1 = 1, tile_dim2 = 0
    g1[5] = (int)nelem8;                             // tensor_dim0_stride[31:0]
    g1[6] = 0;                                       // stride0[47:32]=0, stride1[15:0]=0
    g1[7] = 0;                                       // stride1[47:16]=0
    i32x4 z4 = {0, 0, 0, 0};
    i32x8 z8 = {0, 0, 0, 0, 0, 0, 0, 0};
    __builtin_amdgcn_tensor_load_to_lds(g0, g1, z4, z4, z8, 0);
}
#endif

// ---------------- persistent LSTM kernel ----------------
// grid = 64 WGs x 256 threads. WG bx owns h-columns [bx*16, bx*16+16).
// Wave w (8 per WG) owns batch rows [w*16, w*16+16); holds c-state 16x16 patch in VGPRs.
// All weight B-fragments for this WG (4 gate tiles x full K) live in LDS: 192 KiB.
__global__ void __launch_bounds__(256) lstm_persistent(
        const unsigned short* __restrict__ xbf,   // (T,B,F) bf16
        const unsigned short* __restrict__ wihp,  // packed W_ih fragments
        const unsigned short* __restrict__ whhp,  // packed W_hh fragments
        const float* __restrict__ b_ih, const float* __restrict__ b_hh,
        const float* __restrict__ fc_w, const float* __restrict__ fc_b,
        unsigned short* __restrict__ hbuf,        // 2 * (B,H) bf16, double buffered
        unsigned* __restrict__ ctrl,
        float* __restrict__ out)                  // (B, AOUT) f32
{
    __shared__ unsigned short ldsW[LDS_HALFS];    // 192 KiB

    const int bx   = blockIdx.x;        // 0..63
    const int tid  = threadIdx.x;
    const int lane = tid & 31;
    const int wid  = tid >> 5;          // 0..7
    const int m0   = wid * 16;
    const int kh   = lane >> 4;
    const int mrow = m0 + (lane & 15);  // A-fragment row this lane serves
    const int hc0  = bx * 16;

    // ---- stage this WG's weight fragments into LDS (once) ----
#ifdef HAVE_TDM
    if (wid == 0) {
        // W_hh: per gate, 32 frags * 1 KiB = 32 KiB contiguous (4096 x 8B)
        for (int gi = 0; gi < 4; ++gi) {
            tdm_load_1d(whhp + (size_t)((gi * 64 + bx) * 32) * 512,
                        ldsW + (size_t)gi * 32 * 512, 4096u);
        }
        // W_ih: per gate, 16 frags * 1 KiB = 16 KiB contiguous (2048 x 8B)
        for (int gi = 0; gi < 4; ++gi) {
            tdm_load_1d(wihp + (size_t)((gi * 64 + bx) * 16) * 512,
                        ldsW + (size_t)(LDS_WIH_BASE + gi * 16 * 512), 2048u);
        }
        __builtin_amdgcn_s_wait_tensorcnt(0);
    }
#else
    for (int i = tid; i < 4 * 32 * 32; i += 256) {      // W_hh: i = gate*1024 + kt*32 + lane
        const int gi = i >> 10;
        const int kt = (i >> 5) & 31;
        const int ln = i & 31;
        const uint4* s = (const uint4*)(whhp + ((size_t)((gi * 64 + bx) * 32 + kt) * 32 + ln) * 16);
        uint4* d = (uint4*)(ldsW + ((size_t)(gi * 32 + kt) * 32 + ln) * 16);
        d[0] = s[0]; d[1] = s[1];
    }
    for (int i = tid; i < 4 * 16 * 32; i += 256) {      // W_ih
        const int gi = i >> 9;
        const int kt = (i >> 5) & 15;
        const int ln = i & 31;
        const uint4* s = (const uint4*)(wihp + ((size_t)((gi * 64 + bx) * 16 + kt) * 32 + ln) * 16);
        uint4* d = (uint4*)(ldsW + ((size_t)(LDS_WIH_BASE) + ((size_t)(gi * 16 + kt) * 32 + ln) * 16));
        d[0] = s[0]; d[1] = s[1];
    }
#endif

    // per-lane gate biases for this lane's output column
    const int col = hc0 + (lane & 15);
    float bias[4];
#pragma unroll
    for (int g = 0; g < 4; ++g) bias[g] = b_ih[g * HH + col] + b_hh[g * HH + col];

    v8f cst = {};   // register-resident cell state (16x16 patch, WMMA C layout)
    __syncthreads();

    for (int t = 0; t < TT; ++t) {
        v8f acc[4];
#pragma unroll
        for (int g = 0; g < 4; ++g) acc[g] = (v8f){};

        // prefetch next timestep's x row while computing this one
        if (t + 1 < TT) {
            __builtin_prefetch(xbf + (size_t)(t + 1) * (BB * FF) + (size_t)mrow * FF, 0, 0);
        }

        // ---- input contribution: A from xbf[t] (global), B from LDS ----
        const unsigned short* xr = xbf + (size_t)t * (BB * FF) + (size_t)mrow * FF;
#pragma unroll 4
        for (int kt = 0; kt < FF / 32; ++kt) {
            FragU a;
            a.q[0] = *(const uint4*)(xr + kt * 32 + kh * 8);
            a.q[1] = *(const uint4*)(xr + kt * 32 + 16 + kh * 8);
#pragma unroll
            for (int g = 0; g < 4; ++g) {
                FragU b;
                const uint4* p = (const uint4*)(ldsW + (size_t)LDS_WIH_BASE +
                                                ((size_t)(g * 16 + kt) * 32 + lane) * 16);
                b.q[0] = p[0];
                b.q[1] = p[1];
                acc[g] = __builtin_amdgcn_wmma_f32_16x16x32_bf16(
                    false, a.v, false, b.v, (short)0, acc[g], false, false);
            }
        }

        // ---- recurrent contribution: A from h (global), B from LDS ----
        const unsigned short* hr = hbuf + (size_t)(t & 1) * (BB * HH) + (size_t)mrow * HH;
#pragma unroll 4
        for (int kt = 0; kt < HH / 32; ++kt) {
            FragU a;
            a.q[0] = *(const uint4*)(hr + kt * 32 + kh * 8);
            a.q[1] = *(const uint4*)(hr + kt * 32 + 16 + kh * 8);
#pragma unroll
            for (int g = 0; g < 4; ++g) {
                FragU b;
                const uint4* p = (const uint4*)(ldsW + ((size_t)(g * 32 + kt) * 32 + lane) * 16);
                b.q[0] = p[0];
                b.q[1] = p[1];
                acc[g] = __builtin_amdgcn_wmma_f32_16x16x32_bf16(
                    false, a.v, false, b.v, (short)0, acc[g], false, false);
            }
        }

        // ---- activations + state update + h store (bf16, double buffer) ----
        unsigned short* hw = hbuf + (size_t)((t + 1) & 1) * (BB * HH);
#pragma unroll
        for (int r = 0; r < 8; ++r) {
            float gi = fsigmoid(acc[0][r] + bias[0]);
            float gf = fsigmoid(acc[1][r] + bias[1]);
            float gg = ftanh   (acc[2][r] + bias[2]);
            float go = fsigmoid(acc[3][r] + bias[3]);
            float c  = gf * cst[r] + gi * gg;
            cst[r]   = c;
            float h  = go * ftanh(c);
            const int row = m0 + kh * 8 + r;     // WMMA C/D layout row
            hw[(size_t)row * HH + col] = f2bf(h);
        }

        grid_barrier(&ctrl[0], &ctrl[1]);
    }

    // ---- final FC: out = sigmoid(h_last @ fc_w^T + fc_b), h_last in hbuf[0] (T even) ----
    if (bx < (BB * AOUT + 255) / 256) {
        int idx = bx * 256 + tid;
        if (idx < BB * AOUT) {
            int b = idx / AOUT, a = idx % AOUT;
            const unsigned short* h = hbuf + (size_t)b * HH;
            const float* wr = fc_w + (size_t)a * HH;
            float s = fc_b[a];
            for (int k = 0; k < HH; ++k) s += bf2f(h[k]) * wr[k];
            out[idx] = fsigmoid(s);
        }
    }
}

// ---------------- host launcher ----------------
extern "C" void kernel_launch(void* const* d_in, const int* in_sizes, int n_in,
                              void* d_out, int out_size, void* d_ws, size_t ws_size,
                              hipStream_t stream) {
    const float* x    = (const float*)d_in[0];
    const float* W_ih = (const float*)d_in[1];
    const float* W_hh = (const float*)d_in[2];
    const float* b_ih = (const float*)d_in[3];
    const float* b_hh = (const float*)d_in[4];
    const float* fc_w = (const float*)d_in[5];
    const float* fc_b = (const float*)d_in[6];
    float* out = (float*)d_out;

    char* ws = (char*)d_ws;
    unsigned*       ctrl = (unsigned*)(ws + OFF_CTRL);
    unsigned short* hbuf = (unsigned short*)(ws + OFF_H);
    unsigned short* wihp = (unsigned short*)(ws + OFF_WIH);
    unsigned short* whhp = (unsigned short*)(ws + OFF_WHH);
    unsigned short* xbf  = (unsigned short*)(ws + OFF_XBF);

    // zero barrier state + both h buffers (h0 = 0)
    (void)hipMemsetAsync(ws, 0, OFF_H + (size_t)2 * BB * HH * 2, stream);

    // convert/transpose x -> bf16 (T,B,F)
    {
        size_t n = (size_t)TT * BB * FF;
        prep_x_kernel<<<(unsigned)((n + 255) / 256), 256, 0, stream>>>(x, xbf);
    }
    // pack W_ih and W_hh into WMMA B-fragment order (bf16)
    {
        size_t n = (size_t)G4 * FF + (size_t)G4 * HH;
        pack_w_kernel<<<(unsigned)((n + 255) / 256), 256, 0, stream>>>(W_ih, W_hh, wihp, whhp);
    }
    // persistent recurrence + fused final FC
    lstm_persistent<<<NWG, 256, 0, stream>>>(xbf, wihp, whhp, b_ih, b_hh,
                                             fc_w, fc_b, hbuf, ctrl, out);
    (void)in_sizes; (void)n_in; (void)out_size; (void)ws_size;
}
